// CapsRoute_29970281791747
// MI455X (gfx1250) — matrix operand order repaired and checked by reading, not compiled, not run on hardware
//
#include <hip/hip_runtime.h>

typedef __attribute__((ext_vector_type(16))) _Float16 v16h;
typedef __attribute__((ext_vector_type(8)))  float    v8f;
typedef __attribute__((ext_vector_type(4)))  unsigned int v4u;
typedef __attribute__((ext_vector_type(8)))  int      v8i;
typedef __attribute__((ext_vector_type(4)))  int      v4i;

#define HW      9216      // 96*96
#define WD      96
#define NCH     144
#define NBATCH  8
#define EPS_ROUTE 1e-06f
#define BN_EPS    1e-05f

// LDS patch geometry for the conv kernel: 9 ch x 18 rows x 128 cols (f16)
#define PROWS   18
#define PSTR    128               // padded row stride (halfs, pow2 for cheap decode)
#define PCH     (PROWS * PSTR)    // 2304 halfs per channel
#define ZSLOT   98                // patch[98] is always zero (padding column)

// Fragment-table sizes (halfs)
#define WCF_HALFS (16 * 3 * 32 * 16)     // 24576 per conv
#define WGF_HALFS (16 * 32 * 16)         // 8192
#define WPF_HALFS (16 * 8 * 32 * 16)     // 65536
#define WTAB_HALFS (WGF_HALFS + WPF_HALFS)   // 73728 halfs = 147456 B contiguous

static __device__ __forceinline__ v8f wmma16(v16h a, v16h b, v8f c) {
  // v_wmma_f32_16x16x32_f16 (A,B f16; C/D f32)
  return __builtin_amdgcn_wmma_f32_16x16x32_f16(false, a, false, b, (short)0, c, false, false);
}

// ---------------------------------------------------------------------------
// Prep kernel: f32 weights -> f16 tables laid out exactly as WMMA A-fragments.
// A layout (16-bit A 16x32): lane L holds M=L&15; element e -> K=e+8*(e>>3)+8*(L>>4); conv frags use K=e+16*(L>>4).
// ---------------------------------------------------------------------------
__global__ void prep_weights(const float* __restrict__ wc1, const float* __restrict__ wc2,
                             const float* __restrict__ wg,  const float* __restrict__ wp,
                             _Float16* __restrict__ WcF1, _Float16* __restrict__ WcF2,
                             _Float16* __restrict__ WgF,  _Float16* __restrict__ WpF) {
  const int i0 = blockIdx.x * blockDim.x + threadIdx.x;
  const int stride = gridDim.x * blockDim.x;

  for (int t = i0; t < WCF_HALFS; t += stride) {    // conv A frags [g][s][lane][e]
    const int e = t & 15, lane = (t >> 4) & 31, gs = t >> 9;
    const int s = gs % 3, g = gs / 3;
    const int oc = lane & 15;
    const int K = e + 16 * (lane >> 4);
    const int k = 32 * s + K;
    float v1 = 0.f, v2 = 0.f;
    if (oc < 9 && k < 81) {
      v1 = wc1[(g * 9 + oc) * 81 + k];
      v2 = wc2[(g * 9 + oc) * 81 + k];
    }
    WcF1[t] = (_Float16)v1;
    WcF2[t] = (_Float16)v2;
  }

  for (int t = i0; t < WGF_HALFS; t += stride) {    // gate A frags [k][lane][e]
    const int e = t & 15, lane = (t >> 4) & 31, k = t >> 9;
    const int o = lane & 15;
    const int K = e + 8 * (e >> 3) + 8 * (lane >> 4);
    float v = 0.f;
    if (K < 8) v = wg[(k * 16 + o) * 8 + K];
    WgF[t] = (_Float16)v;
  }

  for (int t = i0; t < WPF_HALFS; t += stride) {    // pose A frags [k][m][lane][e]
    const int e = t & 15, lane = (t >> 4) & 31, km = t >> 9;
    const int m = km & 7, k = km >> 3;
    const int row = 16 * m + (lane & 15);           // (o,q): o=row>>3, q=row&7
    const int o = row >> 3, q = row & 7;
    const int K = e + 8 * (e >> 3) + 8 * (lane >> 4);
    float v = 0.f;
    if (K < 8) v = wp[((k * 16 + o) * 8 + K) * 8 + q];
    WpF[t] = (_Float16)v;
  }
}

// ---------------------------------------------------------------------------
// Grouped 3x3 conv (16 groups, 9 in/9 out ch) + BN + SiLU.
// Implicit-im2col GEMM per group: M=9(pad16), K=81(pad96, 3 WMMA steps), N=16 px.
// Input patch staged in LDS as zero-padded f16; B frags gathered via ds loads.
// ---------------------------------------------------------------------------
template <bool NT_STORE>
__global__ void conv_bn_silu_k(const float* __restrict__ src0, const float* __restrict__ src1,
                               int csplit, int c1n,
                               const _Float16* __restrict__ wfrag,
                               const float* __restrict__ gamma, const float* __restrict__ beta,
                               float* __restrict__ dst) {
  __shared__ _Float16 patch[9 * PCH] __attribute__((aligned(128)));

  const int tid  = threadIdx.x;
  const int lane = tid & 31;
  const int hi   = lane >> 4;
  const int wv   = tid >> 5;

  int blk = blockIdx.x;
  const int rb = blk % 6;  blk /= 6;     // 16-row block
  const int g  = blk % 16; blk /= 16;    // group
  const int b  = blk;                    // batch

  // ---- Stage zero-padded patch: rows rb*16-1..rb*16+16, cols -1..96 ----
  for (int c = 0; c < 9; ++c) {
    const int cg = g * 9 + c;
    const float* sp = (cg < csplit)
        ? (src0 + ((size_t)b * csplit + cg) * HW)
        : (src1 + ((size_t)b * c1n + (cg - csplit)) * HW);
    for (int i = tid; i < PCH; i += 256) {
      const int rr = i >> 7, cc = i & (PSTR - 1);
      const int grow = rb * 16 + rr - 1, gcol = cc - 1;
      float v = 0.f;
      if (cc < 98 && grow >= 0 && grow < WD && gcol >= 0 && gcol < WD)
        v = sp[grow * WD + gcol];
      patch[c * PCH + i] = (_Float16)v;
    }
  }
  __syncthreads();

  // A fragments for this group (one 32B vector load each)
  v16h aa[3];
#pragma unroll
  for (int s = 0; s < 3; ++s)
    aa[s] = *(const v16h*)(wfrag + ((size_t)(g * 3 + s) * 32 + lane) * 16);

  // Hoisted BN params per output row of the C fragment
  float scv[8], bev[8];
#pragma unroll
  for (int r = 0; r < 8; ++r) {
    const int oc = r + 8 * hi;
    if (oc < 9) {
      const int cch = g * 9 + oc;
      scv[r] = gamma[cch] * rsqrtf(1.0f + BN_EPS);
      bev[r] = beta[cch];
    } else { scv[r] = 0.f; bev[r] = 0.f; }
  }

  for (int t = 0; t < 12; ++t) {         // 16 rows x 6 col-tiles / 8 waves
    const int idx  = wv * 12 + t;
    const int rloc = idx / 6;            // 0..15
    const int c0   = (idx % 6) * 16;
    const int row  = rb * 16 + rloc;
    const int lanecol = c0 + (lane & 15);          // patch col base

    v8f acc;
#pragma unroll
    for (int r = 0; r < 8; ++r) acc[r] = 0.f;

#pragma unroll
    for (int s = 0; s < 3; ++s) {
      v16h bfr;
#pragma unroll
      for (int e = 0; e < 16; ++e) {
        // B frag in this code's contiguous order (matches conv A): lane->N, element e -> K = e + 16*(lane>>4)
        const int klo = 32 * s + e;
        const int khi = klo + 16;
        const int icl = klo / 9, rl = klo % 9, khl = rl / 3, kwl = rl % 3;
        const int ich = khi / 9, rh = khi % 9, khh = rh / 3, kwh = rh % 3;
        const bool vl = (klo < 81), vh = (khi < 81);
        // LDS address: ic*PCH + (rloc+kh)*PSTR + (lanecol + kw); ZSLOT is zero.
        const int offL = vl ? (icl * PCH + khl * PSTR + kwl) : -1;
        const int offH = vh ? (ich * PCH + khh * PSTR + kwh) : -1;
        int off;
        if (vl && vh)      off = (hi ? offH : offL) + rloc * PSTR + lanecol;
        else if (vl)       off = hi ? ZSLOT : (offL + rloc * PSTR + lanecol);
        else               off = ZSLOT;
        bfr[e] = patch[off];
      }
      acc = wmma16(aa[s], bfr, acc);
    }

    // Epilogue: BN + SiLU (C layout: M = r + 8*hi, N = lane&15)
#pragma unroll
    for (int r = 0; r < 8; ++r) {
      const int oc = r + 8 * hi;
      if (oc < 9) {
        float v = acc[r] * scv[r] + bev[r];
        v = v / (1.0f + __expf(-v));          // SiLU
        float* p = dst + ((size_t)b * NCH + g * 9 + oc) * HW
                       + (size_t)row * WD + c0 + (lane & 15);
        if (NT_STORE) __builtin_nontemporal_store(v, p);
        else          *p = v;
      }
    }
  }
}

// ---------------------------------------------------------------------------
// Capsule self-routing. Weight-fragment tables (WgF|WpF, 147456B contiguous)
// are DMA'd into LDS by the Tensor Data Mover once per block; all A-fragment
// fetches then come from LDS as ds_load_b128.
// Register budget: no ar[16][8] cache — pass 2 recomputes the gate softmax
// (1 extra WMMA + 8 exp per k) so live state is just outm(64)+asum/inv(16).
// ---------------------------------------------------------------------------
static __device__ __forceinline__ v16h load_pose_frag(const float* __restrict__ y,
                                                      size_t ybase, int k, int px, int hi) {
  // Both lane halves load the same addresses (coalesced); hi half is zeroed.
  float pv[8];
#pragma unroll
  for (int p = 0; p < 8; ++p)
    pv[p] = y[ybase + (size_t)(k * 9 + p) * HW + px];
  v16h bp;
#pragma unroll
  for (int e = 0; e < 8; ++e) bp[e] = hi ? (_Float16)0.0f : (_Float16)pv[e];
#pragma unroll
  for (int e = 8; e < 16; ++e) bp[e] = (_Float16)0.0f;
  return bp;
}

// Gate logits -> softmax weights * act for one input capsule k.
// Returns w[r]*act with o = r + 8*hi at pixel lane&15.
static __device__ __forceinline__ void gate_weights(const float* __restrict__ y,
                                                    const _Float16* __restrict__ WgL,
                                                    const float* __restrict__ bg,
                                                    size_t ybase, int k, int px,
                                                    int lane, int hi, v16h bp,
                                                    float wout[8]) {
  float actv = y[ybase + (size_t)(k * 9 + 8) * HW + px];
  actv = 1.0f / (1.0f + __expf(-actv));

  v16h ag = *(const v16h*)(WgL + ((size_t)k * 32 + lane) * 16);
  v8f zc;
#pragma unroll
  for (int r = 0; r < 8; ++r) zc[r] = 0.f;
  v8f lg = wmma16(ag, bp, zc);

  float l[8], mx = -1e30f;
#pragma unroll
  for (int r = 0; r < 8; ++r) {
    l[r] = lg[r] + bg[k * 16 + (r + 8 * hi)];
    mx = fmaxf(mx, l[r]);
  }
  mx = fmaxf(mx, __shfl_xor(mx, 16, 32));
  float s = 0.f;
#pragma unroll
  for (int r = 0; r < 8; ++r) { l[r] = __expf(l[r] - mx); s += l[r]; }
  s += __shfl_xor(s, 16, 32);
  const float scale = actv / s;
#pragma unroll
  for (int r = 0; r < 8; ++r) wout[r] = l[r] * scale;
}

__global__ void route_k(const float* __restrict__ y,
                        const _Float16* __restrict__ wtab,   // WgF followed by WpF
                        const float* __restrict__ bg, float* __restrict__ z) {
  __shared__ _Float16 lds_w[WTAB_HALFS] __attribute__((aligned(128)));

  const int tid  = threadIdx.x;
  const int lane = tid & 31;
  const int hi   = lane >> 4;
  const int wv   = tid >> 5;

  // ---- TDM: 1-D tensor copy of 18432 x 8B from global into LDS offset 0 ----
  if (wv == 0) {
    const unsigned long long ga = (unsigned long long)(uintptr_t)wtab;
    v4u g0;
    g0[0] = 1u;                                   // count=1 (valid descriptor)
    g0[1] = 0u;                                   // lds_addr = 0 (only LDS object)
    g0[2] = (unsigned)ga;                         // global_addr[31:0]
    g0[3] = (unsigned)((ga >> 32) & 0x1FFFFFFull) | (2u << 30);  // addr[56:32]|type=2
    v8i g1;
    g1[0] = 3 << 16;                              // data_size = 8B
    g1[1] = (int)(18432u << 16);                  // tensor_dim0[15:0]
    g1[2] = 1 << 16;                              // dim0[31:16]=0 | tensor_dim1=1
    g1[3] = (int)(18432u << 16);                  // tile_dim0 = 18432
    g1[4] = 0;                                    // tile_dim1/2 unused
    g1[5] = 18432;                                // tensor_dim0_stride[31:0]
    g1[6] = 0; g1[7] = 0;
    v4i gz; gz[0] = 0; gz[1] = 0; gz[2] = 0; gz[3] = 0;
    v8i gz8; gz8[0] = 0; gz8[1] = 0; gz8[2] = 0; gz8[3] = 0;
    gz8[4] = 0; gz8[5] = 0; gz8[6] = 0; gz8[7] = 0;
    __builtin_amdgcn_tensor_load_to_lds(g0, g1, gz, gz, gz8, 0);
    __builtin_amdgcn_s_wait_tensorcnt(0);
  }

  // The TDM writes LDS through a descriptor, invisible to alias analysis:
  // escape the LDS pointer so loads from lds_w are not folded to undef.
  {
    void* p = (void*)lds_w;
    asm volatile("" : "+v"(p) : : "memory");
  }
  __syncthreads();

  const _Float16* WgL = lds_w;                    // 8192 halfs
  const _Float16* WpL = lds_w + WGF_HALFS;        // 65536 halfs

#pragma unroll 1
  for (int it = 0; it < 9; ++it) {
    const int wid  = (blockIdx.x * 8 + wv) * 9 + it;
    const int b    = wid / 576;                   // 576 tiles per image
    const int pix0 = (wid % 576) * 16;
    const int px   = pix0 + (lane & 15);
    const size_t ybase = (size_t)b * NCH * HW;

    // ---- Pass 1: accumulate ar_sum over k (only 8 live accumulators) ----
    float asum[8];
#pragma unroll
    for (int r = 0; r < 8; ++r) asum[r] = 0.f;

#pragma unroll 1
    for (int k = 0; k < 16; ++k) {
      v16h bp = load_pose_frag(y, ybase, k, px, hi);
      float w[8];
      gate_weights(y, WgL, bg, ybase, k, px, lane, hi, bp, w);
#pragma unroll
      for (int r = 0; r < 8; ++r) asum[r] += w[r];
    }

    float inv[8];
#pragma unroll
    for (int r = 0; r < 8; ++r) { asum[r] += EPS_ROUTE; inv[r] = 1.0f / asum[r]; }

    // ---- Pass 2: recompute gate weights, accumulate pose_out ----
    v8f outm[8];
#pragma unroll
    for (int m = 0; m < 8; ++m)
#pragma unroll
      for (int r = 0; r < 8; ++r) outm[m][r] = 0.f;

#pragma unroll 1
    for (int k = 0; k < 16; ++k) {
      v16h bp = load_pose_frag(y, ybase, k, px, hi);
      float w[8];
      gate_weights(y, WgL, bg, ybase, k, px, lane, hi, bp, w);
      float cf[8];
#pragma unroll
      for (int r = 0; r < 8; ++r) cf[r] = w[r] * inv[r];     // coeff[k][o]

#pragma unroll
      for (int m = 0; m < 8; ++m) {
        v16h ap = *(const v16h*)(WpL + (((size_t)k * 8 + m) * 32 + lane) * 16);
        v8f zc;
#pragma unroll
        for (int r = 0; r < 8; ++r) zc[r] = 0.f;
        v8f u = wmma16(ap, bp, zc);           // rows 16m..16m+15 of (o,q)

        // In M-tile m: lower lanes carry o=2m, upper lanes o=2m+1 (all r)
        float cc;
        if (m < 4) {
          const float oth = __shfl_xor(cf[2 * m + 1], 16, 32);
          cc = hi ? oth : cf[2 * m];
        } else {
          const float oth = __shfl_xor(cf[2 * m - 8], 16, 32);
          cc = hi ? cf[2 * m - 7] : oth;
        }
#pragma unroll
        for (int r = 0; r < 8; ++r) outm[m][r] += cc * u[r];
      }
    }

    // ---- Stores: pose_out -> ch o*9+q, act_out -> ch o*9+8 ----
#pragma unroll
    for (int m = 0; m < 8; ++m) {
      const int o = 2 * m + hi;
#pragma unroll
      for (int r = 0; r < 8; ++r)
        z[ybase + (size_t)(o * 9 + r) * HW + px] = outm[m][r];
    }
#pragma unroll
    for (int r = 0; r < 8; ++r) {
      const int o = r + 8 * hi;
      z[ybase + (size_t)(o * 9 + 8) * HW + px] = asum[r];   // includes EPS
    }
  }
}

// ---------------------------------------------------------------------------
extern "C" void kernel_launch(void* const* d_in, const int* in_sizes, int n_in,
                              void* d_out, int out_size, void* d_ws, size_t ws_size,
                              hipStream_t stream) {
  const float* x0  = (const float*)d_in[0];
  const float* x1  = (const float*)d_in[1];
  const float* wc1 = (const float*)d_in[2];
  const float* g1  = (const float*)d_in[3];
  const float* be1 = (const float*)d_in[4];
  const float* wp  = (const float*)d_in[5];
  const float* wg  = (const float*)d_in[6];
  const float* bgt = (const float*)d_in[7];
  const float* wc2 = (const float*)d_in[8];
  const float* g2  = (const float*)d_in[9];
  const float* be2 = (const float*)d_in[10];

  char* ws = (char*)d_ws;
  const size_t ybytes = (size_t)NBATCH * NCH * HW * sizeof(float);   // 42.5 MB
  float* y = (float*)ws;
  float* z = (float*)(ws + ybytes);
  _Float16* WcF1 = (_Float16*)(ws + 2 * ybytes);
  _Float16* WcF2 = WcF1 + WCF_HALFS;
  _Float16* WgF  = WcF2 + WCF_HALFS;          // WgF and WpF are contiguous
  _Float16* WpF  = WgF  + WGF_HALFS;

  prep_weights<<<96, 256, 0, stream>>>(wc1, wc2, wg, wp, WcF1, WcF2, WgF, WpF);

  // conv_route: input is concat(x0,x1) along channel (split at 72)
  conv_bn_silu_k<false><<<NBATCH * 16 * 6, 256, 0, stream>>>(x0, x1, 72, 72,
                                                             WcF1, g1, be1, y);

  // self-routing: 4608 tiles = 64 blocks x 8 waves x 9 tiles
  route_k<<<64, 256, 0, stream>>>(y, WgF, bgt, z);

  // spagg conv: single source; final output streamed with NT stores
  conv_bn_silu_k<true><<<NBATCH * 16 * 6, 256, 0, stream>>>(z, z, NCH, NCH,
                                                            WcF2, g2, be2,
                                                            (float*)d_out);
}
